// MLPKANLayer_44641890075015
// MI455X (gfx1250) — matrix operand order repaired and checked by reading, not compiled, not run on hardware
//
#include <hip/hip_runtime.h>
#include <hip/hip_bf16.h>

typedef __attribute__((ext_vector_type(16))) _Float16 v16h;
typedef __attribute__((ext_vector_type(8)))  float    v8f;
typedef __attribute__((ext_vector_type(4)))  float    v4f;

#define IN_DIM  128
#define OUT_DIM 128
#define NBLK    8                    // degree+1
#define BATCH   1024
#define KPI     (IN_DIM * NBLK)      // 1024 contraction length per output row
#define HDIM    (OUT_DIM * KPI)      // 131072

// ---------------------------------------------------------------------------
// Pass 1: gather the (statically known) nonzeros of M1 / M2 into compact form.
//   w1c[R] = M1[R, (R%1024)/8]     (one nonzero per row of layer-1)
//   ac[R]  = M2[R/1024, R]         (block-diagonal row of layer-2)
// ---------------------------------------------------------------------------
__global__ __launch_bounds__(256) void kan_compact(const float* __restrict__ M1,
                                                   const float* __restrict__ M2,
                                                   float* __restrict__ w1c,
                                                   float* __restrict__ ac) {
  const int R   = blockIdx.x * 256 + threadIdx.x;        // [0, HDIM)
  const int col = (R & (KPI - 1)) >> 3;                  // (R % 1024) / 8
  w1c[R] = M1[(size_t)R * IN_DIM + col];
  const int i = R >> 10;                                 // R / 1024
  ac[R]  = M2[(size_t)i * HDIM + R];
}

// fast silu: t * 1/(1+exp(-t)), exp via v_exp_f32, div via v_rcp_f32
__device__ __forceinline__ float fast_silu(float t) {
  const float NLOG2E = -1.44269504088896340736f;
  float e = __builtin_amdgcn_exp2f(t * NLOG2E);
  float r = __builtin_amdgcn_rcpf(1.0f + e);
  return t * r;
}

// ---------------------------------------------------------------------------
// Pass 2: fused  out[i,b] = b2[i] + sum_k a[i,k] * silu(w1[i,k]*x[k/8,b] + b1[i,k])
//
// Block = 128 threads (4 waves). Block owns a 16i x 16b output tile
//   (blockIdx.y -> i-group of 16, blockIdx.x -> b-group of 16).
// Wave w reduces K-segment [w*256, w*256+256); cross-wave sum via LDS.
// Per K-chunk of 32 and per i' in [0,16): wave computes B (32x16 f16, silu
// values, each lane producing exactly the 16 values of its B-layout slot),
// builds A = row-i' of a-coeffs (delta trick), and issues
// v_wmma_f32_16x16x32_f16 into an f32 16x16 accumulator.
// ---------------------------------------------------------------------------
__global__ __launch_bounds__(128) void kan_fused(const float* __restrict__ x,
                                                 const float* __restrict__ b1,
                                                 const float* __restrict__ b2,
                                                 const float* __restrict__ w1c,
                                                 const float* __restrict__ ac,
                                                 float* __restrict__ out) {
  __shared__ float red[4 * 256];

  const int lane = threadIdx.x & 31;
  const int wave = threadIdx.x >> 5;
  const int half = lane >> 4;          // 0: K-low half, 1: K-high half
  const int col  = lane & 15;          // b column of B/C/D
  const int row  = lane & 15;          // row of A held by this lane

  const int i0 = blockIdx.y * 16;      // 8 i-groups
  const int b0 = blockIdx.x * 16;      // 64 b-groups
  const int ksegbase = wave * 256;     // 4-way K split

  v8f acc = {};

#pragma unroll 1
  for (int kc = 0; kc < 8; ++kc) {
    const int kbase = ksegbase + kc * 32;      // chunk of 32 k-values
    // B layout: this lane holds K = kbase + 16*half + (0..15)
    const int kb2 = kbase + 16 * half;
    const int jb  = kb2 >> 3;                  // input feature for elems 0..7
    const float xA = x[(size_t)jb * BATCH + b0 + col];        // elems 0..7
    const float xB = x[(size_t)(jb + 1) * BATCH + b0 + col];  // elems 8..15

    // Hoist this lane's A-row coefficients for the whole chunk.
    // A layout: lane holds K = kbase + 16*half + (0..15)
    v16h aAll;
    {
      const float* aRow = ac + (size_t)(i0 + row) * KPI + kbase;
      const int o1 = half * 16, o2 = half * 16 + 8;
      v4f a0 = *(const v4f*)(aRow + o1);
      v4f a1 = *(const v4f*)(aRow + o1 + 4);
      v4f a2 = *(const v4f*)(aRow + o2);
      v4f a3 = *(const v4f*)(aRow + o2 + 4);
#pragma unroll
      for (int p = 0; p < 4; ++p) {
        aAll[p]      = (_Float16)a0[p];
        aAll[4 + p]  = (_Float16)a1[p];
        aAll[8 + p]  = (_Float16)a2[p];
        aAll[12 + p] = (_Float16)a3[p];
      }
    }

#pragma unroll
    for (int ip = 0; ip < 16; ++ip) {
      const size_t rb = (size_t)(i0 + ip) * KPI + kb2;   // 64B aligned
      v4f w0 = *(const v4f*)(w1c + rb);
      v4f w1 = *(const v4f*)(w1c + rb + 4);
      v4f w2 = *(const v4f*)(w1c + rb + 8);
      v4f w3 = *(const v4f*)(w1c + rb + 12);
      v4f c0 = *(const v4f*)(b1 + rb);
      v4f c1 = *(const v4f*)(b1 + rb + 4);
      v4f c2 = *(const v4f*)(b1 + rb + 8);
      v4f c3 = *(const v4f*)(b1 + rb + 12);

      v16h Bv;
#pragma unroll
      for (int p = 0; p < 4; ++p) {
        Bv[p]      = (_Float16)fast_silu(__builtin_fmaf(w0[p], xA, c0[p]));
        Bv[4 + p]  = (_Float16)fast_silu(__builtin_fmaf(w1[p], xA, c1[p]));
        Bv[8 + p]  = (_Float16)fast_silu(__builtin_fmaf(w2[p], xB, c2[p]));
        Bv[12 + p] = (_Float16)fast_silu(__builtin_fmaf(w3[p], xB, c3[p]));
      }

      // delta trick: only row i' of A is nonzero
      const _Float16 sel = (row == ip) ? (_Float16)1.0f : (_Float16)0.0f;
      v16h Av = aAll * sel;

      acc = __builtin_amdgcn_wmma_f32_16x16x32_f16(
          /*neg_a=*/false, Av, /*neg_b=*/false, Bv,
          /*c_mod=*/(short)0, acc, /*reuse_a=*/false, /*reuse_b=*/false);
    }
  }

  // cross-wave K reduction via LDS
  {
    float* dst = &red[wave * 256 + lane * 8];
    *(v4f*)(dst)     = (v4f){acc[0], acc[1], acc[2], acc[3]};
    *(v4f*)(dst + 4) = (v4f){acc[4], acc[5], acc[6], acc[7]};
  }
  __syncthreads();

#pragma unroll
  for (int e = threadIdx.x; e < 256; e += 128) {
    const float s = red[e] + red[256 + e] + red[512 + e] + red[768 + e];
    const int l = e >> 3, v = e & 7;
    const int il = v + 8 * (l >> 4);   // C/D layout: VGPR v, lane half -> M
    const int bl = l & 15;
    out[(size_t)(i0 + il) * BATCH + b0 + bl] = s + b2[i0 + il];
  }
}

extern "C" void kernel_launch(void* const* d_in, const int* in_sizes, int n_in,
                              void* d_out, int out_size, void* d_ws, size_t ws_size,
                              hipStream_t stream) {
  // setup_inputs order: x, M1, mask1, b1, M2, mask2, b2  (masks unused: static structure)
  const float* x  = (const float*)d_in[0];
  const float* M1 = (const float*)d_in[1];
  const float* b1 = (const float*)d_in[3];
  const float* M2 = (const float*)d_in[4];
  const float* b2 = (const float*)d_in[6];
  float* out = (float*)d_out;

  float* w1c = (float*)d_ws;       // HDIM floats
  float* ac  = w1c + HDIM;         // HDIM floats  (1 MB total)

  kan_compact<<<HDIM / 256, 256, 0, stream>>>(M1, M2, w1c, ac);

  dim3 grid(BATCH / 16, OUT_DIM / 16);   // (64, 8)
  kan_fused<<<grid, 128, 0, stream>>>(x, b1, b2, w1c, ac, out);
}
